// TreeLSTMCell_63153199121098
// MI455X (gfx1250) — matrix operand order repaired and checked by reading, not compiled, not run on hardware
//
#include <hip/hip_runtime.h>

typedef __bf16 bf16;
typedef __attribute__((ext_vector_type(16))) __bf16 v16bf;
typedef __attribute__((ext_vector_type(8)))  __bf16 v8bf;
typedef __attribute__((ext_vector_type(8)))  float  v8f;
typedef __attribute__((ext_vector_type(4)))  float  f32x4;

#define N_NODES 400000
#define H 128
#define X 128
#define TWO_H 256
#define THREE_H 384
#define NODES_PER_BLOCK 128
#define THREADS 256

// d_ws layout (bf16 elements): Uf[256*256] | Uiou[384*256] | Wiou[384*128]
#define WS_UF    0
#define WS_UIOU  (256*256)
#define WS_WIOU  (256*256 + 384*256)
#define WS_TOTAL (256*256 + 384*256 + 384*128)

// ---------------- weight prep: f32 -> bf16 (runs once per launch) ----------
__global__ void prep_weights(const float* __restrict__ W_iou,
                             const float* __restrict__ U_iou,
                             const float* __restrict__ U_f,
                             bf16* __restrict__ ws) {
  int i = blockIdx.x * blockDim.x + threadIdx.x;
  int stride = gridDim.x * blockDim.x;
  for (; i < WS_TOTAL; i += stride) {
    float v;
    if (i < WS_UIOU)      v = U_f[i];
    else if (i < WS_WIOU) v = U_iou[i - WS_UIOU];
    else                  v = W_iou[i - WS_WIOU];
    ws[i] = (bf16)v;
  }
}

// ---------------- fragment helpers ----------------------------------------
__device__ __forceinline__ v16bf cat16(v8bf lo, v8bf hi) {
  return __builtin_shufflevector(lo, hi, 0,1,2,3,4,5,6,7,8,9,10,11,12,13,14,15);
}

// A (16x32 bf16) fragment from a contiguous bf16 row in LDS.
// lanes 0-15 (khalf=0): K = kb+{0..7, 16..23}; lanes 16-31: K = kb+{8..15, 24..31}
__device__ __forceinline__ v16bf load_a(const bf16* row, int kb, int khalf) {
  const v8bf* p0 = (const v8bf*)(row + kb + khalf * 8);
  const v8bf* p1 = (const v8bf*)(row + kb + 16 + khalf * 8);
  return cat16(*p0, *p1);
}

// B (32x16 bf16) fragment = transpose-view of weight W [rows x ldk] row-major.
// lane's output col n = n0 + (lane&15) -> W row; K = kb+{0..7,16..23} (+8 for lanes 16-31).
__device__ __forceinline__ v16bf load_b(const bf16* __restrict__ W, int ldk,
                                        int n0, int kb, int nlo, int khalf) {
  const bf16* p = W + (size_t)(n0 + nlo) * ldk + kb + khalf * 8;
  const v8bf* p0 = (const v8bf*)p;
  const v8bf* p1 = (const v8bf*)(p + 16);
  return cat16(*p0, *p1);
}

// v_rcp_f32-based activations: accuracy far beyond bf16-GEMM noise floor.
__device__ __forceinline__ float fast_sig(float v) {
  return __builtin_amdgcn_rcpf(1.0f + __expf(-v));
}
__device__ __forceinline__ float fast_tanh(float v) {
  float e = __expf(2.0f * v);
  return (e - 1.0f) * __builtin_amdgcn_rcpf(e + 1.0f);
}

#define WMMA_BF16(A, B, C) \
  __builtin_amdgcn_wmma_f32_16x16x32_bf16(false, (A), false, (B), (short)0, (C), false, false)

// ---------------- fused TreeLSTM cell --------------------------------------
__global__ __launch_bounds__(THREADS) void treelstm_kernel(
    const float* __restrict__ x, const float* __restrict__ h,
    const float* __restrict__ c, const int* __restrict__ children,
    const int* __restrict__ is_leaf,
    const float* __restrict__ b_Wiou, const float* __restrict__ b_Uiou,
    const float* __restrict__ b_Uf,
    const bf16* __restrict__ ws,
    float* __restrict__ out_h, float* __restrict__ out_c) {
  extern __shared__ char smem[];
  bf16* sh_hcat = (bf16*)smem;                                            // [128][256] bf16
  bf16* sh_x    = (bf16*)(smem + NODES_PER_BLOCK * TWO_H * sizeof(bf16)); // [128][128] bf16

  const int tid  = threadIdx.x;
  const int base = blockIdx.x * NODES_PER_BLOCK;

  // ---- stage: gather h[children] -> bf16 h_cat, and x -> bf16 (LDS) ----
  {
    const int i = tid >> 1, j = tid & 1;   // 128 nodes x 2 slots
    const int node = base + i;
    const int ch = children[node * 2 + j];
    const f32x4* src = (const f32x4*)(h + (size_t)ch * H);   // cached: rows shared
    bf16* dst = sh_hcat + i * TWO_H + j * H;
#pragma unroll
    for (int q = 0; q < H / 4; ++q) {
      f32x4 v = src[q];
      dst[q * 4 + 0] = (bf16)v[0]; dst[q * 4 + 1] = (bf16)v[1];
      dst[q * 4 + 2] = (bf16)v[2]; dst[q * 4 + 3] = (bf16)v[3];
    }
    const f32x4* sx = (const f32x4*)(x + (size_t)node * X + j * (X / 2));
    bf16* dx = sh_x + i * X + j * (X / 2);
#pragma unroll
    for (int q = 0; q < X / 8; ++q) {
      f32x4 v = __builtin_nontemporal_load(sx + q);          // x read exactly once
      dx[q * 4 + 0] = (bf16)v[0]; dx[q * 4 + 1] = (bf16)v[1];
      dx[q * 4 + 2] = (bf16)v[2]; dx[q * 4 + 3] = (bf16)v[3];
    }
  }
  __syncthreads();

  const int lane  = tid & 31;
  const int wave  = tid >> 5;
  const int nlo   = lane & 15;   // output-col / A-row selector within tile
  const int khalf = lane >> 4;   // K-half selector
  const int nodeBase = base + wave * 16;

  const bf16* aRow  = sh_hcat + (wave * 16 + nlo) * TWO_H;
  const bf16* axRow = sh_x    + (wave * 16 + nlo) * X;
  const bf16* Uf   = ws + WS_UF;
  const bf16* Uiou = ws + WS_UIOU;
  const bf16* Wiou = ws + WS_WIOU;

  // Per-lane epilogue metadata: D-matrix rows m = d + 8*khalf (d = vgpr idx).
  int ch0[8], ch1[8], lf[8];
#pragma unroll
  for (int d = 0; d < 8; ++d) {
    int node = nodeBase + d + 8 * khalf;
    ch0[d] = children[node * 2];
    ch1[d] = children[node * 2 + 1];
    lf[d]  = is_leaf[node];
  }

#pragma unroll 1
  for (int t = 0; t < 8; ++t) {            // 8 column tiles of H=128
    v8f f0 = {}, f1 = {};                  // forget gates, child 0 / child 1
    v8f hi = {}, ho = {}, hu = {};         // U_iou path (internal nodes)
    v8f xi = {}, xo = {}, xu = {};         // W_iou path (leaf nodes)

#pragma unroll
    for (int ks = 0; ks < TWO_H / 32; ++ks) {   // K over h_cat (256)
      const int kb = ks * 32;
      v16bf a = load_a(aRow, kb, khalf);
      f0 = WMMA_BF16(a, load_b(Uf,   TWO_H,  16 * t,         kb, nlo, khalf), f0);
      f1 = WMMA_BF16(a, load_b(Uf,   TWO_H,  H + 16 * t,     kb, nlo, khalf), f1);
      hi = WMMA_BF16(a, load_b(Uiou, TWO_H,  16 * t,         kb, nlo, khalf), hi);
      ho = WMMA_BF16(a, load_b(Uiou, TWO_H,  H + 16 * t,     kb, nlo, khalf), ho);
      hu = WMMA_BF16(a, load_b(Uiou, TWO_H,  2 * H + 16 * t, kb, nlo, khalf), hu);
    }
#pragma unroll
    for (int ks = 0; ks < X / 32; ++ks) {       // K over x (128)
      const int kb = ks * 32;
      v16bf a = load_a(axRow, kb, khalf);
      xi = WMMA_BF16(a, load_b(Wiou, X, 16 * t,         kb, nlo, khalf), xi);
      xo = WMMA_BF16(a, load_b(Wiou, X, H + 16 * t,     kb, nlo, khalf), xo);
      xu = WMMA_BF16(a, load_b(Wiou, X, 2 * H + 16 * t, kb, nlo, khalf), xu);
    }

    // ---- epilogue for columns [16t, 16t+16) ----
    const int col = 16 * t + nlo;
    const float bUf0 = b_Uf[col],   bUf1 = b_Uf[H + col];
    const float bWi  = b_Wiou[col], bWo  = b_Wiou[H + col], bWu = b_Wiou[2 * H + col];
    const float bUi  = b_Uiou[col], bUo  = b_Uiou[H + col], bUu = b_Uiou[2 * H + col];

#pragma unroll
    for (int d = 0; d < 8; ++d) {
      const int node = nodeBase + d + 8 * khalf;
      const float c0 = c[(size_t)ch0[d] * H + col];
      const float c1 = c[(size_t)ch1[d] * H + col];
      const float fg0 = fast_sig(f0[d] + bUf0);
      const float fg1 = fast_sig(f1[d] + bUf1);
      const float cf = fg0 * c0 + fg1 * c1;
      float iv, ov, uv;
      if (lf[d]) { iv = xi[d] + bWi; ov = xo[d] + bWo; uv = xu[d] + bWu; }
      else       { iv = hi[d] + bUi; ov = ho[d] + bUo; uv = hu[d] + bUu; }
      const float cn = fast_sig(iv) * fast_tanh(uv) + cf;
      const float hn = fast_sig(ov) * fast_tanh(cn);
      __builtin_nontemporal_store(hn, &out_h[(size_t)node * H + col]);
      __builtin_nontemporal_store(cn, &out_c[(size_t)node * H + col]);
    }
  }
}

// ---------------- launch ----------------------------------------------------
extern "C" void kernel_launch(void* const* d_in, const int* in_sizes, int n_in,
                              void* d_out, int out_size, void* d_ws, size_t ws_size,
                              hipStream_t stream) {
  const float* x        = (const float*)d_in[0];
  const float* h        = (const float*)d_in[1];
  const float* c        = (const float*)d_in[2];
  const int*   children = (const int*)d_in[3];
  const int*   is_leaf  = (const int*)d_in[4];
  const float* W_iou    = (const float*)d_in[5];
  const float* b_Wiou   = (const float*)d_in[6];
  const float* U_iou    = (const float*)d_in[7];
  const float* b_Uiou   = (const float*)d_in[8];
  const float* U_f      = (const float*)d_in[9];
  const float* b_Uf     = (const float*)d_in[10];

  bf16*  ws    = (bf16*)d_ws;
  float* out_h = (float*)d_out;
  float* out_c = out_h + (size_t)N_NODES * H;

  prep_weights<<<(WS_TOTAL + 255) / 256, 256, 0, stream>>>(W_iou, U_iou, U_f, ws);

  const size_t smem = (size_t)NODES_PER_BLOCK * (TWO_H + X) * sizeof(bf16); // 96 KB
  treelstm_kernel<<<N_NODES / NODES_PER_BLOCK, THREADS, smem, stream>>>(
      x, h, c, children, is_leaf, b_Wiou, b_Uiou, b_Uf, ws, out_h, out_c);
}
